// VQAE_89970974917370
// MI455X (gfx1250) — matrix operand order repaired and checked by reading, not compiled, not run on hardware
//
#include <hip/hip_runtime.h>
#include <math.h>

// ---- problem constants (from reference) ----
#define NROWS 131072
#define DIN   256
#define HDIM  64
#define DZ    32
#define KCB   1024
#define BNEPS 1e-5f

typedef __attribute__((ext_vector_type(16))) _Float16 v16h;
typedef __attribute__((ext_vector_type(8)))  float    v8f;

// D = A(16x32 f16) * B(32x16 f16) + C(16x16 f32)
__device__ __forceinline__ v8f wmma16(v16h a, v16h b, v8f c) {
  return __builtin_amdgcn_wmma_f32_16x16x32_f16(
      false, a, false, b, (short)0, c, false, false);
}

// A-matrix 16-bit 16x32 layout (contiguous, matching B):
// lanes 0-15: halves 0..15 -> K=0..15
// lanes 16-31: halves 0..15 -> K=16..31
__device__ __forceinline__ int akof(int lane, int h) {
  return ((lane & 16) ? 16 : 0) + h;
}
// B-matrix 32x16: lane = column (lane&15); lanes<16 hold K=0..15 in halves,
// lanes>=16 hold K=16..31.
// C/D f32: value i -> M = i + (lane&16 ? 8 : 0), N = lane & 15.

// ---- CDNA5 async HBM->LDS copy (bypasses VGPRs; tracked by ASYNCcnt) ----
// VDST VGPR = LDS byte address (low 32 bits of flat pointer), VADDR = 64-bit
// global address, no SADDR. Completion fenced with s_wait_asynccnt.
__device__ __forceinline__ void async_to_lds_b128(unsigned lds_addr, const void* g) {
  asm volatile("global_load_async_to_lds_b128 %0, %1, off"
               :: "v"(lds_addr), "v"((unsigned long long)(uintptr_t)g)
               : "memory");
}
__device__ __forceinline__ void wait_async0() {
  asm volatile("s_wait_asynccnt 0x0" ::: "memory");
}

// =====================================================================
// kernel 0: zero global accumulators
// =====================================================================
__global__ void k_init(float* gs1, float* gq1, float* gs2, float* gq2, float* accs) {
  int t = threadIdx.x;
  if (t < 64) { gs1[t] = 0.f; gq1[t] = 0.f; gs2[t] = 0.f; gq2[t] = 0.f; }
  if (t < 2)  accs[t] = 0.f;
}

// =====================================================================
// kernel 1: h1 = X @ W1 + b1   [N,64], accumulate per-column sum/sumsq
// block: 256 thr (8 waves), each wave: 16 rows x 64 cols; grid 1024
// X tile staged HBM->LDS via global_load_async_to_lds_b128 (f32, pitched),
// f32->f16 conversion deferred to the A-fragment build.
// =====================================================================
__global__ void k_enc_l1(const float* __restrict__ X, const float* __restrict__ W1,
                         const float* __restrict__ b1, float* __restrict__ hbuf,
                         float* __restrict__ gsum, float* __restrict__ gsq) {
  __shared__ _Float16 sWT[HDIM * 272];            // W1^T [col][k], pitched
  __shared__ __align__(16) float sXf[128 * 36];   // X tile f32 [row][32], pitched
  __shared__ float sCs[HDIM], sCq[HDIM];
  const int tid = threadIdx.x, wave = tid >> 5, lane = tid & 31;
  if (tid < HDIM) { sCs[tid] = 0.f; sCq[tid] = 0.f; }
  for (int i = tid; i < HDIM * DIN; i += 256) {
    int n = i & 63, k = i >> 6;
    sWT[n * 272 + k] = (_Float16)W1[k * HDIM + n];
  }
  __syncthreads();
  const int rowBase = blockIdx.x * 128;
  v8f acc[4] = {};
  for (int kt = 0; kt < 8; ++kt) {
    __syncthreads();   // previous tile fully consumed before overwrite
    // 128 rows x 32 floats = 1024 x b128 transfers, 4 per thread
#pragma unroll
    for (int t = 0; t < 4; ++t) {
      const int tr  = tid + t * 256;
      const int r   = tr >> 3, seg = tr & 7;
      const unsigned loff = (unsigned)(uintptr_t)&sXf[r * 36 + seg * 4];
      async_to_lds_b128(loff, &X[(size_t)(rowBase + r) * DIN + kt * 32 + seg * 4]);
    }
    wait_async0();
    __syncthreads();
    v16h a;
    const int rl = wave * 16 + (lane & 15);
#pragma unroll
    for (int h = 0; h < 16; ++h) a[h] = (_Float16)sXf[rl * 36 + akof(lane, h)];
#pragma unroll
    for (int nt = 0; nt < 4; ++nt) {
      v16h b;
      const int col = nt * 16 + (lane & 15);
      const int kb  = kt * 32 + ((lane & 16) ? 16 : 0);
#pragma unroll
      for (int h = 0; h < 16; ++h) b[h] = sWT[col * 272 + kb + h];
      acc[nt] = wmma16(a, b, acc[nt]);
    }
  }
  const int mBase = rowBase + wave * 16 + ((lane & 16) ? 8 : 0);
#pragma unroll
  for (int nt = 0; nt < 4; ++nt) {
    const int col = nt * 16 + (lane & 15);
    const float bias = b1[col];
    float s = 0.f, s2 = 0.f;
#pragma unroll
    for (int i = 0; i < 8; ++i) {
      float v = acc[nt][i] + bias;
      hbuf[(size_t)(mBase + i) * HDIM + col] = v;
      s += v; s2 += v * v;
    }
    atomicAdd(&sCs[col], s);
    atomicAdd(&sCq[col], s2);
  }
  __syncthreads();
  if (tid < HDIM) { atomicAdd(&gsum[tid], sCs[tid]); atomicAdd(&gsq[tid], sCq[tid]); }
}

// =====================================================================
// kernel 2/5: BN finalize: scale = g*rsqrt(var+eps), shift = beta - mean*scale
// =====================================================================
__global__ void k_bn_fin(const float* __restrict__ sum, const float* __restrict__ sq,
                         const float* __restrict__ g, const float* __restrict__ be,
                         float* __restrict__ scale, float* __restrict__ shift) {
  int t = threadIdx.x;
  if (t < HDIM) {
    float mean = sum[t] * (1.0f / (float)NROWS);
    float var  = sq[t] * (1.0f / (float)NROWS) - mean * mean;
    float sc   = g[t] * rsqrtf(var + BNEPS);
    scale[t] = sc;
    shift[t] = be[t] - mean * sc;
  }
}

// =====================================================================
// kernel 3: a1 = relu(BN(h1)); z = a1 @ W2 + b2; VQ argmin over 1024
// codes (WMMA distance GEMM); topics + gathered exact min-dist sum.
// =====================================================================
__global__ void k_enc_vq(const float* __restrict__ hbuf,
                         const float* __restrict__ scaleA, const float* __restrict__ shiftA,
                         const float* __restrict__ W2, const float* __restrict__ b2,
                         const float* __restrict__ cb,
                         int* __restrict__ topics, float* __restrict__ tout,
                         float* __restrict__ minsumAcc) {
  __shared__ _Float16 sCB[KCB * 40];     // codebook f16 [code][32], pitched
  __shared__ float    sCN[KCB];          // ||c||^2
  __shared__ _Float16 sW2T[DZ * 72];     // W2^T [col][k]
  __shared__ float    sZ[128 * DZ];      // z f32 per block tile
  __shared__ float    sZN[128];          // ||z||^2
  __shared__ float    sScale[HDIM], sShift[HDIM], sB2[DZ];
  __shared__ float    sLoss;
  const int tid = threadIdx.x, wave = tid >> 5, lane = tid & 31;
  if (tid == 0) sLoss = 0.f;
  if (tid < HDIM) { sScale[tid] = scaleA[tid]; sShift[tid] = shiftA[tid]; }
  if (tid < DZ) sB2[tid] = b2[tid];
  for (int i = tid; i < DZ * HDIM; i += 256) {
    int n = i & 31, k = i >> 5;
    sW2T[n * 72 + k] = (_Float16)W2[k * DZ + n];
  }
  for (int c = tid; c < KCB; c += 256) {
    float s = 0.f;
    for (int j = 0; j < DZ; ++j) {
      float v = cb[c * DZ + j];
      s += v * v;
      sCB[c * 40 + j] = (_Float16)v;
    }
    sCN[c] = s;
  }
  __syncthreads();

  const int rowBase = blockIdx.x * 128 + wave * 16;
  const int rl = rowBase + (lane & 15);
  const int mB = (lane & 16) ? 8 : 0;

  // ---- z = relu(BN(h1)) @ W2 + b2  (2 K-steps x 2 N-tiles of WMMA) ----
  v16h a0, a1;
#pragma unroll
  for (int h = 0; h < 16; ++h) {
    int k0 = akof(lane, h);
    float v0 = hbuf[(size_t)rl * HDIM + k0] * sScale[k0] + sShift[k0];
    a0[h] = (_Float16)(v0 > 0.f ? v0 : 0.f);
    int k1 = 32 + k0;
    float v1 = hbuf[(size_t)rl * HDIM + k1] * sScale[k1] + sShift[k1];
    a1[h] = (_Float16)(v1 > 0.f ? v1 : 0.f);
  }
#pragma unroll
  for (int nt = 0; nt < 2; ++nt) {
    const int col = nt * 16 + (lane & 15);
    const int kb  = (lane & 16) ? 16 : 0;
    v8f c = {};
    v16h b;
#pragma unroll
    for (int h = 0; h < 16; ++h) b[h] = sW2T[col * 72 + kb + h];
    c = wmma16(a0, b, c);
#pragma unroll
    for (int h = 0; h < 16; ++h) b[h] = sW2T[col * 72 + 32 + kb + h];
    c = wmma16(a1, b, c);
    const float bias = sB2[col];
#pragma unroll
    for (int i = 0; i < 8; ++i)
      sZ[(wave * 16 + mB + i) * DZ + col] = c[i] + bias;
  }
  __syncthreads();
  if (lane < 16) {
    float s = 0.f;
    for (int j = 0; j < DZ; ++j) {
      float v = sZ[(wave * 16 + lane) * DZ + j];
      s += v * v;
    }
    sZN[wave * 16 + lane] = s;
  }
  __syncthreads();

  // ---- distance GEMM: one WMMA per 16-code chunk, K=32 ----
  v16h za;
#pragma unroll
  for (int h = 0; h < 16; ++h)
    za[h] = (_Float16)sZ[(wave * 16 + (lane & 15)) * DZ + akof(lane, h)];
  float zn[8];
#pragma unroll
  for (int i = 0; i < 8; ++i) zn[i] = sZN[wave * 16 + mB + i];

  float mind[8]; int midx[8];
#pragma unroll
  for (int i = 0; i < 8; ++i) { mind[i] = 3.4e38f; midx[i] = 0; }

  const int kb2 = (lane & 16) ? 16 : 0;
  for (int chunk = 0; chunk < KCB / 16; ++chunk) {
    const int code = chunk * 16 + (lane & 15);
    v16h b;
#pragma unroll
    for (int h = 0; h < 16; ++h) b[h] = sCB[code * 40 + kb2 + h];
    v8f d = {};
    d = wmma16(za, b, d);
    const float cn = sCN[code];
#pragma unroll
    for (int i = 0; i < 8; ++i) {
      float d2 = zn[i] + cn - 2.0f * d[i];
      if (d2 < mind[i]) { mind[i] = d2; midx[i] = code; }
    }
  }
  // reduce across the 16 lanes of each half (N dimension), lowest-index tie-break
#pragma unroll
  for (int m = 1; m < 16; m <<= 1) {
#pragma unroll
    for (int i = 0; i < 8; ++i) {
      float od = __shfl_xor(mind[i], m, 32);
      int   oi = __shfl_xor(midx[i], m, 32);
      if (od < mind[i] || (od == mind[i] && oi < midx[i])) { mind[i] = od; midx[i] = oi; }
    }
  }
  // lanes 0..7 own rows 0..7, lanes 16..23 own rows 8..15
  if ((lane & 15) < 8) {
    const int i = lane & 15;
    const int row = rowBase + mB + i;
    const int t = midx[i];
    topics[row] = t;
    tout[row] = (float)t;
    float s = 0.f;
    for (int j = 0; j < DZ; ++j) {
      float diff = sZ[(wave * 16 + mB + i) * DZ + j] - cb[t * DZ + j];
      s += diff * diff;
    }
    atomicAdd(&sLoss, s);
  }
  __syncthreads();
  if (tid == 0) atomicAdd(minsumAcc, sLoss);
}

// =====================================================================
// kernel 4: h2 = codebook[topics] @ dec_W1 + b1, accumulate BN stats
// =====================================================================
__global__ void k_dec_l1(const int* __restrict__ topics, const float* __restrict__ cb,
                         const float* __restrict__ W1, const float* __restrict__ b1,
                         float* __restrict__ hbuf,
                         float* __restrict__ gsum, float* __restrict__ gsq) {
  __shared__ _Float16 sCB[KCB * 40];
  __shared__ _Float16 sWT[HDIM * 72];    // dec_W1^T [col][k<32]
  __shared__ float sCs[HDIM], sCq[HDIM];
  const int tid = threadIdx.x, wave = tid >> 5, lane = tid & 31;
  if (tid < HDIM) { sCs[tid] = 0.f; sCq[tid] = 0.f; }
  for (int c = tid; c < KCB; c += 256)
    for (int j = 0; j < DZ; ++j) sCB[c * 40 + j] = (_Float16)cb[c * DZ + j];
  for (int i = tid; i < HDIM * DZ; i += 256) {
    int n = i & 63, k = i >> 6;
    sWT[n * 72 + k] = (_Float16)W1[k * HDIM + n];
  }
  __syncthreads();
  const int row = blockIdx.x * 128 + wave * 16 + (lane & 15);
  const int t = topics[row];
  v16h a;
#pragma unroll
  for (int h = 0; h < 16; ++h) a[h] = sCB[t * 40 + akof(lane, h)];
  v8f acc[4] = {};
#pragma unroll
  for (int nt = 0; nt < 4; ++nt) {
    const int col = nt * 16 + (lane & 15);
    const int kb  = (lane & 16) ? 16 : 0;
    v16h b;
#pragma unroll
    for (int h = 0; h < 16; ++h) b[h] = sWT[col * 72 + kb + h];
    acc[nt] = wmma16(a, b, acc[nt]);
  }
  const int mBase = blockIdx.x * 128 + wave * 16 + ((lane & 16) ? 8 : 0);
#pragma unroll
  for (int nt = 0; nt < 4; ++nt) {
    const int col = nt * 16 + (lane & 15);
    const float bias = b1[col];
    float s = 0.f, s2 = 0.f;
#pragma unroll
    for (int i = 0; i < 8; ++i) {
      float v = acc[nt][i] + bias;
      hbuf[(size_t)(mBase + i) * HDIM + col] = v;
      s += v; s2 += v * v;
    }
    atomicAdd(&sCs[col], s);
    atomicAdd(&sCq[col], s2);
  }
  __syncthreads();
  if (tid < HDIM) { atomicAdd(&gsum[tid], sCs[tid]); atomicAdd(&gsq[tid], sCq[tid]); }
}

// =====================================================================
// kernel 6: X_rec = relu(BN(h2)) @ dec_W2 + b2; accumulate SSE vs X
// =====================================================================
__global__ void k_dec_l2(const float* __restrict__ hbuf,
                         const float* __restrict__ scaleB, const float* __restrict__ shiftB,
                         const float* __restrict__ W2, const float* __restrict__ b2,
                         const float* __restrict__ X, float* __restrict__ sseAcc) {
  __shared__ _Float16 sWT[DIN * 72];     // dec_W2^T [col][k<64]
  __shared__ float sB[DIN], sScale[HDIM], sShift[HDIM];
  __shared__ float sSse;
  const int tid = threadIdx.x, wave = tid >> 5, lane = tid & 31;
  if (tid == 0) sSse = 0.f;
  if (tid < HDIM) { sScale[tid] = scaleB[tid]; sShift[tid] = shiftB[tid]; }
  for (int i = tid; i < DIN; i += 256) sB[i] = b2[i];
  for (int i = tid; i < DIN * HDIM; i += 256) {
    int n = i & 255, k = i >> 8;
    sWT[n * 72 + k] = (_Float16)W2[k * DIN + n];
  }
  __syncthreads();
  const int rl = blockIdx.x * 128 + wave * 16 + (lane & 15);
  v16h a0, a1;
#pragma unroll
  for (int h = 0; h < 16; ++h) {
    int k0 = akof(lane, h);
    float v0 = hbuf[(size_t)rl * HDIM + k0] * sScale[k0] + sShift[k0];
    a0[h] = (_Float16)(v0 > 0.f ? v0 : 0.f);
    int k1 = 32 + k0;
    float v1 = hbuf[(size_t)rl * HDIM + k1] * sScale[k1] + sShift[k1];
    a1[h] = (_Float16)(v1 > 0.f ? v1 : 0.f);
  }
  const int mBase = blockIdx.x * 128 + wave * 16 + ((lane & 16) ? 8 : 0);
  const int kb = (lane & 16) ? 16 : 0;
  float sse = 0.f;
  for (int nt = 0; nt < 16; ++nt) {
    const int col = nt * 16 + (lane & 15);
    v8f c = {};
    v16h b;
#pragma unroll
    for (int h = 0; h < 16; ++h) b[h] = sWT[col * 72 + kb + h];
    c = wmma16(a0, b, c);
#pragma unroll
    for (int h = 0; h < 16; ++h) b[h] = sWT[col * 72 + 32 + kb + h];
    c = wmma16(a1, b, c);
    const float bias = sB[col];
#pragma unroll
    for (int i = 0; i < 8; ++i) {
      float v = c[i] + bias - X[(size_t)(mBase + i) * DIN + col];
      sse += v * v;
    }
  }
  atomicAdd(&sSse, sse);
  __syncthreads();
  if (tid == 0) atomicAdd(sseAcc, sSse);
}

// =====================================================================
// kernel 7: loss = 2*minsum + sqrt(sse)
// =====================================================================
__global__ void k_fin(const float* __restrict__ accs, float* __restrict__ out) {
  if (threadIdx.x == 0) out[0] = 2.0f * accs[0] + sqrtf(accs[1]);
}

// =====================================================================
extern "C" void kernel_launch(void* const* d_in, const int* in_sizes, int n_in,
                              void* d_out, int out_size, void* d_ws, size_t ws_size,
                              hipStream_t stream) {
  const float* X    = (const float*)d_in[0];
  const float* eW1  = (const float*)d_in[1];
  const float* eb1  = (const float*)d_in[2];
  const float* eg1  = (const float*)d_in[3];
  const float* ebe1 = (const float*)d_in[4];
  const float* eW2  = (const float*)d_in[5];
  const float* eb2  = (const float*)d_in[6];
  const float* cb   = (const float*)d_in[7];
  const float* dW1  = (const float*)d_in[8];
  const float* db1  = (const float*)d_in[9];
  const float* dg1  = (const float*)d_in[10];
  const float* dbe1 = (const float*)d_in[11];
  const float* dW2  = (const float*)d_in[12];
  const float* db2  = (const float*)d_in[13];
  float* out = (float*)d_out;  // [0] = loss, [1..N] = topics as float

  // workspace layout (~34.1 MB)
  float* ws     = (float*)d_ws;
  float* hbuf   = ws;                               // N*64 floats (reused enc/dec)
  float* gsum1  = hbuf + (size_t)NROWS * HDIM;      // 64
  float* gsq1   = gsum1 + 64;
  float* gsum2  = gsq1 + 64;
  float* gsq2   = gsum2 + 64;
  float* scaleA = gsq2 + 64;
  float* shiftA = scaleA + 64;
  float* scaleB = shiftA + 64;
  float* shiftB = scaleB + 64;
  float* accs   = shiftB + 64;                      // [0]=minsum, [1]=sse
  int*   topics = (int*)(accs + 2);                 // N ints

  const int grid = NROWS / 128;  // 1024 blocks, 256 threads (8 wave32)

  k_init<<<1, 64, 0, stream>>>(gsum1, gsq1, gsum2, gsq2, accs);
  k_enc_l1<<<grid, 256, 0, stream>>>(X, eW1, eb1, hbuf, gsum1, gsq1);
  k_bn_fin<<<1, 64, 0, stream>>>(gsum1, gsq1, eg1, ebe1, scaleA, shiftA);
  k_enc_vq<<<grid, 256, 0, stream>>>(hbuf, scaleA, shiftA, eW2, eb2, cb,
                                     topics, out + 1, &accs[0]);
  k_dec_l1<<<grid, 256, 0, stream>>>(topics, cb, dW1, db1, hbuf, gsum2, gsq2);
  k_bn_fin<<<1, 64, 0, stream>>>(gsum2, gsq2, dg1, dbe1, scaleB, shiftB);
  k_dec_l2<<<grid, 256, 0, stream>>>(hbuf, scaleB, shiftB, dW2, db2, X, &accs[1]);
  k_fin<<<1, 1, 0, stream>>>(accs, out);
}
